// GroupedExperts_68736656605440
// MI455X (gfx1250) — matrix operand order repaired and checked by reading, not compiled, not run on hardware
//
#include <hip/hip_runtime.h>
#include <cstdint>

// ---------------------------------------------------------------------------
// Grouped MoE FFN for MI455X (gfx1250, wave32, WMMA).
// zero/count -> prefix -> scatter -> x->bf16 -> grouped GEMM1 (x@Wgu, fused
// silu(g)*u, bf16 h) -> grouped GEMM2 (h@Wd, weighted atomic scatter).
// Core math: v_wmma_f32_16x16x32_bf16. A tiles staged global->LDS with
// global_load_async_to_lds_b128 (ASYNCcnt); B tiles loaded as global_load_b128
// (forced AS(1) via ext-vector), converted fp32->bf16 in 32-bit VALU and
// stored transposed as packed b32 into padded LDS.
// ---------------------------------------------------------------------------

typedef unsigned short u16;
typedef u16   u16x8   __attribute__((ext_vector_type(8)));
typedef u16   u16x16  __attribute__((ext_vector_type(16)));
typedef __bf16 bf16x16 __attribute__((ext_vector_type(16)));
typedef float v8f     __attribute__((ext_vector_type(8)));
typedef float f32x4   __attribute__((ext_vector_type(4)));

namespace {
constexpr int E = 8, D = 2048, I = 1408, K = 2, T = 4096;
constexpr int TM = 128, TN = 128, KC = 32;
constexpr int LDSS = 40;            // padded LDS row stride (elements, 80B)
}

__device__ __forceinline__ u16 f2bf(float f) {          // fp32 -> bf16 (RNE)
  unsigned u = __float_as_uint(f);
  u += 0x7FFFu + ((u >> 16) & 1u);
  return (u16)(u >> 16);
}

// Pack two fp32 -> {bf16(lo), bf16(hi)} in one u32, all 32-bit VALU ops
// (avoids 16-bit sub-register movs).
__device__ __forceinline__ unsigned pk2(float lo, float hi) {
  unsigned ul = __float_as_uint(lo);
  unsigned uh = __float_as_uint(hi);
  ul += 0x7FFFu + ((ul >> 16) & 1u);
  uh += 0x7FFFu + ((uh >> 16) & 1u);
  return (ul >> 16) | (uh & 0xFFFF0000u);
}

// Force a true global (addrspace 1) 128-bit load: global_load_b128.
// Uses a trivial ext-vector type so the AS(1) lvalue-to-rvalue load needs no
// constructor (HIP's float4 class cannot bind an AS(1) reference).
__device__ __forceinline__ f32x4 gload4(const float* p) {
#if defined(__HIP_DEVICE_COMPILE__)
  typedef __attribute__((address_space(1))) const f32x4* gp;
  return *(gp)(uintptr_t)p;
#else
  return *(const f32x4*)p;
#endif
}

// Generic LDS pointer: low 32 bits are the LDS byte address (aperture map).
__device__ __forceinline__ unsigned lds_off(const void* p) {
  return (unsigned)(uintptr_t)p;
}

// 16B global -> LDS async copy, tracked with ASYNCcnt (cdna5 §08).
__device__ __forceinline__ void async_copy_b128(unsigned lds_addr, const void* gptr) {
  asm volatile("global_load_async_to_lds_b128 %0, %1, off"
               :: "v"(lds_addr), "v"(gptr) : "memory");
}
__device__ __forceinline__ void wait_async0() {
  asm volatile("s_wait_asynccnt 0" ::: "memory");
}

// Two contiguous 16B LDS chunks -> one 16-lane bf16 WMMA fragment.
__device__ __forceinline__ bf16x16 ldfrag(const u16* p0, const u16* p1) {
  u16x8 lo = *(const u16x8*)p0;
  u16x8 hi = *(const u16x8*)p1;
  u16x16 t = __builtin_shufflevector(lo, hi, 0,1,2,3,4,5,6,7,8,9,10,11,12,13,14,15);
  return __builtin_bit_cast(bf16x16, t);
}

// ---------------------------------------------------------------- utilities
__global__ void zero_out_k(float4* __restrict__ out4, int n4, int* __restrict__ ctrs) {
  int i = blockIdx.x * blockDim.x + threadIdx.x;
  float4 z; z.x = z.y = z.z = z.w = 0.f;
  for (; i < n4; i += gridDim.x * blockDim.x) out4[i] = z;
  if (blockIdx.x == 0 && threadIdx.x < 16) ctrs[threadIdx.x] = 0;   // counts[8]+fill[8]
}

__global__ void count_k(const int* __restrict__ idx, int* __restrict__ counts) {
  int i = blockIdx.x * blockDim.x + threadIdx.x;
  if (i < T * K) {
    int e = idx[i]; e = e < 0 ? 0 : (e >= E ? E - 1 : e);
    atomicAdd(&counts[e], 1);
  }
}

__global__ void prefix_k(const int* __restrict__ counts, int* __restrict__ offs) {
  if (blockIdx.x == 0 && threadIdx.x == 0) {
    int s = 0;
    for (int e = 0; e < E; ++e) { offs[e] = s; s += counts[e]; }
    offs[E] = s;
  }
}

__global__ void scatter_k(const int* __restrict__ idx, const float* __restrict__ w,
                          const unsigned char* __restrict__ mask,
                          const int* __restrict__ offs, int* __restrict__ fill,
                          int* __restrict__ rows, float* __restrict__ wts) {
  int i = blockIdx.x * blockDim.x + threadIdx.x;
  if (i < T * K) {
    int t = i / K;
    int e = idx[i]; e = e < 0 ? 0 : (e >= E ? E - 1 : e);
    int pos = offs[e] + atomicAdd(&fill[e], 1);
    rows[pos] = t;
    wts[pos]  = mask[t] ? w[i] : 0.f;
  }
}

__global__ void cvt_x_k(const float* __restrict__ x, uint2* __restrict__ xb2, int n4) {
  int i = blockIdx.x * blockDim.x + threadIdx.x;
  if (i < n4) {
    f32x4 v = gload4(x + (size_t)i * 4);
    uint2 o; o.x = pk2(v.x, v.y); o.y = pk2(v.z, v.w);
    xb2[i] = o;
  }
}

// ------------------------------------------------- GEMM1: gu = x @ Wgu, silu
__global__ __launch_bounds__(256) void gemm1_gu(
    const u16* __restrict__ xb, const float* __restrict__ guw,
    const int* __restrict__ rows, const int* __restrict__ offs,
    u16* __restrict__ hb) {
  const int e = blockIdx.z, nt = blockIdx.x, mt = blockIdx.y;
  const int p0  = offs[e];
  const int cnt = offs[e + 1] - p0;
  if (mt * TM >= cnt) return;
  const int pbase = p0 + mt * TM;
  const int mrem  = cnt - mt * TM;

  __shared__ u16 sA [TM * LDSS];
  __shared__ u16 sBg[TN * LDSS];
  __shared__ u16 sBu[TN * LDSS];
  __shared__ int sRow[TM];

  const int tid = threadIdx.x;
  const int wave = tid >> 5, lane = tid & 31;
  const int wm = wave >> 2, wn = wave & 3;       // 2x4 wave grid: 64x32 each
  const int hlf = lane >> 4, l15 = lane & 15;

  if (tid < TM) {
    int t2 = 0;
    if (tid < mrem) { t2 = rows[pbase + tid]; t2 = t2 < 0 ? 0 : (t2 >= T ? T - 1 : t2); }
    sRow[tid] = t2;
  }
  __syncthreads();

  const float* Wg = guw + (size_t)e * D * (2 * I) + (size_t)nt * TN;

  // ---- hoisted per-thread staging addresses (k-invariant) ----
  // A: thread async-copies chunks (tid) and (tid+256): 8 bf16 = 16B each.
  const int am0 = tid >> 2, ack = tid & 3;
  const int am1 = (tid + 256) >> 2;
  const u16* ap0 = xb + (size_t)sRow[am0] * D + ack * 8;
  const u16* ap1 = xb + (size_t)sRow[am1] * D + ack * 8;
  const unsigned al0 = lds_off(&sA[am0 * LDSS + ack * 8]);
  const unsigned al1 = lds_off(&sA[am1 * LDSS + ack * 8]);
  // B: thread owns unit (k-pair kp, column quad n0) for j=0,1.
  const float* bp[2]; u16* sgb[2]; u16* sub[2];
  #pragma unroll
  for (int j = 0; j < 2; ++j) {
    int uu = tid + j * 256;
    int kp = uu >> 5, n0 = (uu & 31) * 4;
    bp[j]  = Wg + (size_t)(kp * 2) * (2 * I) + n0;
    sgb[j] = &sBg[n0 * LDSS + kp * 2];
    sub[j] = &sBu[n0 * LDSS + kp * 2];
  }

  v8f z8;
  #pragma unroll
  for (int q = 0; q < 8; ++q) z8[q] = 0.f;
  v8f accg[4][2], accu[4][2];
  #pragma unroll
  for (int mi = 0; mi < 4; ++mi)
    #pragma unroll
    for (int ni = 0; ni < 2; ++ni) { accg[mi][ni] = z8; accu[mi][ni] = z8; }

  for (int k0 = 0; k0 < D; k0 += KC) {
    // A tile: async global->LDS (no VGPR round trip), overlaps B conversion.
    async_copy_b128(al0, ap0);
    async_copy_b128(al1, ap1);
    ap0 += KC; ap1 += KC;
    // B gate+up tiles: global_load_b128, fp32 -> bf16, transposed to [n][k].
    #pragma unroll
    for (int j = 0; j < 2; ++j) {
      const float* p = bp[j];
      f32x4 g0 = gload4(p);
      f32x4 g1 = gload4(p + 2 * I);
      f32x4 q0 = gload4(p + I);
      f32x4 q1 = gload4(p + I + 2 * I);
      u16* sg = sgb[j]; u16* su = sub[j];
      *(unsigned*)(sg + 0 * LDSS) = pk2(g0.x, g1.x);
      *(unsigned*)(sg + 1 * LDSS) = pk2(g0.y, g1.y);
      *(unsigned*)(sg + 2 * LDSS) = pk2(g0.z, g1.z);
      *(unsigned*)(sg + 3 * LDSS) = pk2(g0.w, g1.w);
      *(unsigned*)(su + 0 * LDSS) = pk2(q0.x, q1.x);
      *(unsigned*)(su + 1 * LDSS) = pk2(q0.y, q1.y);
      *(unsigned*)(su + 2 * LDSS) = pk2(q0.z, q1.z);
      *(unsigned*)(su + 3 * LDSS) = pk2(q0.w, q1.w);
      if (k0 + KC < D) __builtin_prefetch(p + (size_t)KC * (2 * I), 0, 1);
      bp[j] = p + (size_t)KC * (2 * I);
    }
    wait_async0();
    __syncthreads();

    bf16x16 af[4];
    #pragma unroll
    for (int mi = 0; mi < 4; ++mi) {
      int r = wm * 64 + mi * 16 + l15;
      const u16* p = &sA[r * LDSS + hlf * 16];
      af[mi] = ldfrag(p, p + 8);
    }
    bf16x16 bg[2], bu[2];
    #pragma unroll
    for (int ni = 0; ni < 2; ++ni) {
      int n = wn * 32 + ni * 16 + l15;
      const u16* pg = &sBg[n * LDSS + hlf * 16];
      const u16* pu = &sBu[n * LDSS + hlf * 16];
      bg[ni] = ldfrag(pg, pg + 8);
      bu[ni] = ldfrag(pu, pu + 8);
    }
    #pragma unroll
    for (int mi = 0; mi < 4; ++mi)
      #pragma unroll
      for (int ni = 0; ni < 2; ++ni) {
        accg[mi][ni] = __builtin_amdgcn_wmma_f32_16x16x32_bf16(
            false, af[mi], false, bg[ni], (short)0, accg[mi][ni], false, false);
        accu[mi][ni] = __builtin_amdgcn_wmma_f32_16x16x32_bf16(
            false, af[mi], false, bu[ni], (short)0, accu[mi][ni], false, false);
      }
    __syncthreads();
  }

  // Epilogue: h = silu(gate) * up, stored bf16.
  #pragma unroll
  for (int mi = 0; mi < 4; ++mi)
    #pragma unroll
    for (int ni = 0; ni < 2; ++ni) {
      int col = nt * TN + wn * 32 + ni * 16 + l15;
      #pragma unroll
      for (int r = 0; r < 8; ++r) {
        int m = wm * 64 + mi * 16 + hlf * 8 + r;
        if (m < mrem) {
          float g = accg[mi][ni][r], u = accu[mi][ni][r];
          float s = g / (1.f + __expf(-g));
          hb[(size_t)(pbase + m) * I + col] = f2bf(s * u);
        }
      }
    }
}

// ------------------------------------------- GEMM2: y = h @ Wd, scatter out
__global__ __launch_bounds__(256) void gemm2_down(
    const u16* __restrict__ hb, const float* __restrict__ dnw,
    const int* __restrict__ rows, const float* __restrict__ wts,
    const int* __restrict__ offs, float* __restrict__ out) {
  const int e = blockIdx.z, nt = blockIdx.x, mt = blockIdx.y;
  const int p0  = offs[e];
  const int cnt = offs[e + 1] - p0;
  if (mt * TM >= cnt) return;
  const int pbase = p0 + mt * TM;
  const int mrem  = cnt - mt * TM;

  __shared__ u16 sA[TM * LDSS];
  __shared__ u16 sB[TN * LDSS];
  __shared__ int   sRow[TM];
  __shared__ float sW[TM];

  const int tid = threadIdx.x;
  const int wave = tid >> 5, lane = tid & 31;
  const int wm = wave >> 2, wn = wave & 3;
  const int hlf = lane >> 4, l15 = lane & 15;

  if (tid < TM) {
    int t2 = 0; float w = 0.f;
    if (tid < mrem) {
      t2 = rows[pbase + tid]; t2 = t2 < 0 ? 0 : (t2 >= T ? T - 1 : t2);
      w  = wts[pbase + tid];
    }
    sRow[tid] = t2; sW[tid] = w;
  }
  __syncthreads();

  const float* Wd = dnw + (size_t)e * I * D + (size_t)nt * TN;

  // Hoisted staging addresses.
  const int am0 = tid >> 2, ack = tid & 3;
  const int am1 = (tid + 256) >> 2;
  const u16* ap0 = hb + (size_t)(pbase + am0) * I + ack * 8;
  const u16* ap1 = hb + (size_t)(pbase + am1) * I + ack * 8;
  const unsigned al0 = lds_off(&sA[am0 * LDSS + ack * 8]);
  const unsigned al1 = lds_off(&sA[am1 * LDSS + ack * 8]);
  const float* bp[2]; u16* sbb[2];
  #pragma unroll
  for (int j = 0; j < 2; ++j) {
    int uu = tid + j * 256;
    int kp = uu >> 5, n0 = (uu & 31) * 4;
    bp[j]  = Wd + (size_t)(kp * 2) * D + n0;
    sbb[j] = &sB[n0 * LDSS + kp * 2];
  }

  v8f z8;
  #pragma unroll
  for (int q = 0; q < 8; ++q) z8[q] = 0.f;
  v8f acc[4][2];
  #pragma unroll
  for (int mi = 0; mi < 4; ++mi)
    #pragma unroll
    for (int ni = 0; ni < 2; ++ni) acc[mi][ni] = z8;

  for (int k0 = 0; k0 < I; k0 += KC) {
    async_copy_b128(al0, ap0);
    async_copy_b128(al1, ap1);
    ap0 += KC; ap1 += KC;
    #pragma unroll
    for (int j = 0; j < 2; ++j) {
      const float* p = bp[j];
      f32x4 b0 = gload4(p);
      f32x4 b1 = gload4(p + D);
      u16* sb = sbb[j];
      *(unsigned*)(sb + 0 * LDSS) = pk2(b0.x, b1.x);
      *(unsigned*)(sb + 1 * LDSS) = pk2(b0.y, b1.y);
      *(unsigned*)(sb + 2 * LDSS) = pk2(b0.z, b1.z);
      *(unsigned*)(sb + 3 * LDSS) = pk2(b0.w, b1.w);
      if (k0 + KC < I) __builtin_prefetch(p + (size_t)KC * D, 0, 1);
      bp[j] = p + (size_t)KC * D;
    }
    wait_async0();
    __syncthreads();

    bf16x16 af[4];
    #pragma unroll
    for (int mi = 0; mi < 4; ++mi) {
      int r = wm * 64 + mi * 16 + l15;
      const u16* p = &sA[r * LDSS + hlf * 16];
      af[mi] = ldfrag(p, p + 8);
    }
    bf16x16 bfr[2];
    #pragma unroll
    for (int ni = 0; ni < 2; ++ni) {
      int n = wn * 32 + ni * 16 + l15;
      const u16* pb = &sB[n * LDSS + hlf * 16];
      bfr[ni] = ldfrag(pb, pb + 8);
    }
    #pragma unroll
    for (int mi = 0; mi < 4; ++mi)
      #pragma unroll
      for (int ni = 0; ni < 2; ++ni)
        acc[mi][ni] = __builtin_amdgcn_wmma_f32_16x16x32_bf16(
            false, af[mi], false, bfr[ni], (short)0, acc[mi][ni], false, false);
    __syncthreads();
  }

  // Epilogue: out[token] += routing_weight * y (each token has exactly K=2 hits).
  #pragma unroll
  for (int mi = 0; mi < 4; ++mi)
    #pragma unroll
    for (int ni = 0; ni < 2; ++ni) {
      int col = nt * TN + wn * 32 + ni * 16 + l15;
      #pragma unroll
      for (int r = 0; r < 8; ++r) {
        int m = wm * 64 + mi * 16 + hlf * 8 + r;
        if (m < mrem) {
          float y = acc[mi][ni][r] * sW[m];
          atomicAdd(&out[(size_t)sRow[m] * D + col], y);
        }
      }
    }
}

// --------------------------------------------------------------------- host
extern "C" void kernel_launch(void* const* d_in, const int* in_sizes, int n_in,
                              void* d_out, int out_size, void* d_ws, size_t ws_size,
                              hipStream_t stream) {
  (void)in_sizes; (void)n_in; (void)out_size; (void)ws_size;
  const float*         x    = (const float*)d_in[0];
  const unsigned char* mask = (const unsigned char*)d_in[1];
  const float*         w    = (const float*)d_in[2];
  const int*           idx  = (const int*)d_in[3];
  const float*         guw  = (const float*)d_in[4];
  const float*         dnw  = (const float*)d_in[5];
  float* out = (float*)d_out;

  char* base = (char*)d_ws;
  size_t off = 0;
  auto carve = [&](size_t bytes) -> char* {
    char* p = base + off;
    off = (off + bytes + 255) & ~(size_t)255;
    return p;
  };
  int*   ctrs   = (int*)  carve(16 * sizeof(int));              // counts[8], fill[8]
  int*   counts = ctrs;
  int*   fill   = ctrs + 8;
  int*   offs   = (int*)  carve((E + 1) * sizeof(int));
  int*   rows   = (int*)  carve((size_t)T * K * sizeof(int));
  float* wts    = (float*)carve((size_t)T * K * sizeof(float));
  u16*   xb     = (u16*)  carve((size_t)T * D * sizeof(u16));            // x in bf16
  u16*   hb     = (u16*)  carve((size_t)(T * K + TM) * I * sizeof(u16)); // h (+tile slack)

  zero_out_k<<<4096, 256, 0, stream>>>((float4*)out, T * D / 4, ctrs);
  count_k  <<<(T * K + 255) / 256, 256, 0, stream>>>(idx, counts);
  prefix_k <<<1, 1, 0, stream>>>(counts, offs);
  scatter_k<<<(T * K + 255) / 256, 256, 0, stream>>>(idx, w, mask, offs, fill, rows, wts);
  cvt_x_k  <<<(T * D / 4 + 255) / 256, 256, 0, stream>>>(x, (uint2*)xb, T * D / 4);

  gemm1_gu  <<<dim3(I / TN, (T * K + TM - 1) / TM, E), 256, 0, stream>>>(xb, guw, rows, offs, hb);
  gemm2_down<<<dim3(D / TN, (T * K + TM - 1) / TM, E), 256, 0, stream>>>(hb, dnw, rows, wts, offs, out);
}
